// cnn_lstm_283467842420
// MI455X (gfx1250) — matrix operand order repaired and checked by reading, not compiled, not run on hardware
//
#include <hip/hip_runtime.h>
#include <math.h>

// ---------------------------------------------------------------------------
// CNN+2-layer LSTM (H=2048, B=32, T=4) for MI455X (gfx1250, wave32, WMMA).
//
// Strategy (see analysis): bandwidth-bound recurrent GEMMs; convert the three
// 8192x2048 weights to bf16 once (96 MB, L2-resident on MI455X's 192MB L2),
// precompute layer-0 input GEMM (K=19) + biases, then 8 sequential WMMA cell
// kernels (stream order = recurrence dependency). v_wmma_f32_16x16x32_bf16
// with f32 accumulation over K=2048 (or 4096 for layer 1).
// ---------------------------------------------------------------------------

#define H      2048
#define FOURH  8192
#define BATCH  32
#define TSTEPS 4
#define NW     (FOURH * H)   // elements per big weight matrix

typedef __attribute__((ext_vector_type(16))) __bf16 bf16x16;
typedef __attribute__((ext_vector_type(8)))  __bf16 bf16x8;
typedef __attribute__((ext_vector_type(4)))  __bf16 bf16x4;
typedef __attribute__((ext_vector_type(8)))  float  f32x8;

// ---------------- small prep kernels ----------------

// Zero a region (u32 granularity).
__global__ void k_zero(unsigned int* __restrict__ p, int n) {
    int i = blockIdx.x * blockDim.x + threadIdx.x;
    if (i < n) p[i] = 0u;
}

// fp32 -> bf16 weight conversion, 4 elements per thread (b128 load / b64 store).
__global__ void k_cvt4(const float* __restrict__ s, __bf16* __restrict__ d, int n4) {
    int i = blockIdx.x * blockDim.x + threadIdx.x;
    int stride = gridDim.x * blockDim.x;
    for (; i < n4; i += stride) {
        float4 v = ((const float4*)s)[i];
        bf16x4 o;
        o[0] = (__bf16)v.x; o[1] = (__bf16)v.y;
        o[2] = (__bf16)v.z; o[3] = (__bf16)v.w;
        ((bf16x4*)d)[i] = o;
    }
}

// x = tanh(conv_w * concat(state,action) + conv_b), laid out (t*32+b)*19 + f.
__global__ void k_xt(const float* __restrict__ state, const float* __restrict__ action,
                     const float* __restrict__ cw, const float* __restrict__ cb,
                     float* __restrict__ xt) {
    int id = blockIdx.x * blockDim.x + threadIdx.x;
    if (id >= TSTEPS * BATCH * 19) return;
    int f  = id % 19;
    int tb = id / 19;          // t*32 + b
    int b  = tb & 31;
    int t  = tb >> 5;
    float xv = (f < 15) ? state[(b * TSTEPS + t) * 15 + f]
                        : action[(b * TSTEPS + t) * 4 + (f - 15)];
    xt[id] = tanhf(cw[0] * xv + cb[0]);
}

// pre0[(t*32+b)*8192 + g] = bias0[g] + sum_f x[t,b,f] * w_ih0[g,f]   (K=19)
__global__ void k_pre0(const float* __restrict__ xt, const float* __restrict__ w_ih0,
                       const float* __restrict__ b_ih0, const float* __restrict__ b_hh0,
                       float* __restrict__ pre0) {
    int id = blockIdx.x * blockDim.x + threadIdx.x;
    if (id >= TSTEPS * BATCH * FOURH) return;
    int g  = id & (FOURH - 1);
    int tb = id >> 13;
    const float* x = xt + tb * 19;
    const float* w = w_ih0 + g * 19;
    float s = b_ih0[g] + b_hh0[g];
    #pragma unroll
    for (int f = 0; f < 19; ++f) s += x[f] * w[f];
    pre0[id] = s;
}

// pre1[b*8192 + g] = b_ih1[g] + b_hh1[g]
__global__ void k_pre1(const float* __restrict__ b_ih1, const float* __restrict__ b_hh1,
                       float* __restrict__ pre1) {
    int id = blockIdx.x * blockDim.x + threadIdx.x;
    if (id >= BATCH * FOURH) return;
    pre1[id] = b_ih1[id & (FOURH - 1)] + b_hh1[id & (FOURH - 1)];
}

// ---------------- main WMMA LSTM-cell kernel ----------------
//
// One (layer, timestep). Grid: 128 blocks (16 H-columns each) x 128 threads
// (4 waves; wave g owns gate g). Per wave: 2 M-tiles (rows 0-15 / 16-31),
// K-loop of v_wmma_f32_16x16x32_bf16. Gates combined via 8KB LDS, then the
// LSTM nonlinearity updates c and writes h (f32 optional + bf16 ping-pong).
__global__ __launch_bounds__(128)
void k_cell(const __bf16* __restrict__ A1, const __bf16* __restrict__ W1,
            const __bf16* __restrict__ A2, const __bf16* __restrict__ W2,
            const float*  __restrict__ pre, float* __restrict__ c,
            float* __restrict__ hf, __bf16* __restrict__ hb) {
    __shared__ float gsm[4][BATCH * 16];

    const int nb   = blockIdx.x * 16;     // H-column group base
    const int gate = threadIdx.x >> 5;    // 0..3 : i, f, g, o
    const int lane = threadIdx.x & 31;
    const int nn   = lane & 15;
    const int half = lane >> 4;

    f32x8 acc0 = {};   // rows 0..15
    f32x8 acc1 = {};   // rows 16..31

    const __bf16* As[2] = {A1, A2};
    const __bf16* Ws[2] = {W1, W2};
    for (int pass = 0; pass < 2; ++pass) {
        const __bf16* A = As[pass];
        if (A == nullptr) continue;
        // B-matrix column n == weight row (gate*H + nb + nn), contiguous in K.
        const __bf16* Wrow = Ws[pass] + (size_t)(gate * H + nb + nn) * H;
        for (int kb = 0; kb < H; kb += 32) {
            // A fragment (16-bit 16x32 layout): lane holds K = {half*8..+7}
            // in comps 0..7 and K = {16+half*8..+7} in comps 8..15.
            bf16x16 a0, a1;
            {
                const __bf16* pa = A + nn * H + kb + half * 8;
                bf16x8 lo = *(const bf16x8*)pa;
                bf16x8 hi = *(const bf16x8*)(pa + 16);
                #pragma unroll
                for (int j = 0; j < 8; ++j) { a0[j] = lo[j]; a0[8 + j] = hi[j]; }
            }
            {
                const __bf16* pa = A + (16 + nn) * H + kb + half * 8;
                bf16x8 lo = *(const bf16x8*)pa;
                bf16x8 hi = *(const bf16x8*)(pa + 16);
                #pragma unroll
                for (int j = 0; j < 8; ++j) { a1[j] = lo[j]; a1[8 + j] = hi[j]; }
            }
            // B fragment: same K order as A (half*8+j, then 16+half*8+j).
            bf16x16 bm;
            {
                const __bf16* pb = Wrow + kb + half * 8;
                bf16x8 lo = *(const bf16x8*)pb;
                bf16x8 hi = *(const bf16x8*)(pb + 16);
                #pragma unroll
                for (int j = 0; j < 8; ++j) { bm[j] = lo[j]; bm[8 + j] = hi[j]; }
            }
            acc0 = __builtin_amdgcn_wmma_f32_16x16x32_bf16(
                false, a0, false, bm, (short)0, acc0, false, false);
            acc1 = __builtin_amdgcn_wmma_f32_16x16x32_bf16(
                false, a1, false, bm, (short)0, acc1, false, false);
        }
    }

    // D layout: component r -> row half*8 + r (per M-tile), column nn.
    #pragma unroll
    for (int r = 0; r < 8; ++r) {
        gsm[gate][(half * 8 + r) * 16 + nn]      = acc0[r];
        gsm[gate][(16 + half * 8 + r) * 16 + nn] = acc1[r];
    }
    __syncthreads();

    // LSTM cell: each of 128 threads handles 4 (row, col) positions, coalesced.
    #pragma unroll
    for (int e = 0; e < 4; ++e) {
        int p   = e * 128 + threadIdx.x;     // 0..511 over (32 rows x 16 cols)
        int row = p >> 4;
        int col = nb + (p & 15);
        float gi_ = gsm[0][p] + pre[row * FOURH + col];
        float gf_ = gsm[1][p] + pre[row * FOURH + H + col];
        float gg_ = gsm[2][p] + pre[row * FOURH + 2 * H + col];
        float go_ = gsm[3][p] + pre[row * FOURH + 3 * H + col];
        float iv = 1.0f / (1.0f + expf(-gi_));
        float fv = 1.0f / (1.0f + expf(-gf_));
        float gv = tanhf(gg_);
        float ov = 1.0f / (1.0f + expf(-go_));
        float cold = c[row * H + col];
        float cnew = fv * cold + iv * gv;
        float hv   = ov * tanhf(cnew);
        c[row * H + col]  = cnew;
        hb[row * H + col] = (__bf16)hv;
        if (hf) hf[row * H + col] = hv;
    }
}

// out[b] = sigmoid( sum_{t,n} h1f[(t*32+b)*H + n] * lin_w[t*H + n] + lin_b )
__global__ __launch_bounds__(256)
void k_final(const float* __restrict__ h1f, const float* __restrict__ lin_w,
             const float* __restrict__ lin_b, float* __restrict__ out) {
    __shared__ float red[256];
    int b = blockIdx.x;
    float s = 0.0f;
    for (int i = threadIdx.x; i < FOURH; i += 256) {
        int t = i >> 11;
        int n = i & (H - 1);
        s += h1f[(t * BATCH + b) * H + n] * lin_w[i];
    }
    red[threadIdx.x] = s;
    __syncthreads();
    for (int st = 128; st > 0; st >>= 1) {
        if (threadIdx.x < st) red[threadIdx.x] += red[threadIdx.x + st];
        __syncthreads();
    }
    if (threadIdx.x == 0) out[b] = 1.0f / (1.0f + expf(-(red[0] + lin_b[0])));
}

// ---------------- host orchestration ----------------

extern "C" void kernel_launch(void* const* d_in, const int* in_sizes, int n_in,
                              void* d_out, int out_size, void* d_ws, size_t ws_size,
                              hipStream_t stream) {
    const float* state  = (const float*)d_in[0];
    const float* action = (const float*)d_in[1];
    const float* conv_w = (const float*)d_in[2];
    const float* conv_b = (const float*)d_in[3];
    const float* w_ih0  = (const float*)d_in[4];
    const float* w_hh0  = (const float*)d_in[5];
    const float* b_ih0  = (const float*)d_in[6];
    const float* b_hh0  = (const float*)d_in[7];
    const float* w_ih1  = (const float*)d_in[8];
    const float* w_hh1  = (const float*)d_in[9];
    const float* b_ih1  = (const float*)d_in[10];
    const float* b_hh1  = (const float*)d_in[11];
    const float* lin_w  = (const float*)d_in[12];
    const float* lin_b  = (const float*)d_in[13];
    float* out = (float*)d_out;

    // Workspace carve-up (~108 MB total; all offsets 256B aligned).
    char* ws = (char*)d_ws;
    size_t off = 0;
    auto alloc = [&](size_t bytes) {
        char* p = ws + off;
        off += (bytes + 255) & ~(size_t)255;
        return p;
    };
    __bf16* wbf_hh0 = (__bf16*)alloc((size_t)NW * 2);
    __bf16* wbf_ih1 = (__bf16*)alloc((size_t)NW * 2);
    __bf16* wbf_hh1 = (__bf16*)alloc((size_t)NW * 2);
    float*  pre0    = (float*) alloc((size_t)TSTEPS * BATCH * FOURH * 4);
    float*  pre1    = (float*) alloc((size_t)BATCH * FOURH * 4);
    float*  xt      = (float*) alloc((size_t)TSTEPS * BATCH * 19 * 4);
    // Contiguous zero-region: h0bf[2], h1bf[2], c0, c1  (1 MB)
    char*   zstart  = ws + off;
    __bf16* h0bf    = (__bf16*)alloc((size_t)2 * BATCH * H * 2);
    __bf16* h1bf    = (__bf16*)alloc((size_t)2 * BATCH * H * 2);
    float*  c0      = (float*) alloc((size_t)BATCH * H * 4);
    float*  c1      = (float*) alloc((size_t)BATCH * H * 4);
    size_t  zbytes  = (size_t)((ws + off) - zstart);
    float*  h1f     = (float*) alloc((size_t)TSTEPS * BATCH * H * 4);

    // 1) zero h/c state (bf16 zero == 0x0000)
    {
        int n = (int)(zbytes / 4);
        k_zero<<<(n + 255) / 256, 256, 0, stream>>>((unsigned int*)zstart, n);
    }
    // 2) bf16 weight conversion (grid-stride, float4->bf16x4)
    k_cvt4<<<2048, 256, 0, stream>>>(w_hh0, wbf_hh0, NW / 4);
    k_cvt4<<<2048, 256, 0, stream>>>(w_ih1, wbf_ih1, NW / 4);
    k_cvt4<<<2048, 256, 0, stream>>>(w_hh1, wbf_hh1, NW / 4);
    // 3) conv+tanh, layer-0 input GEMM (K=19), layer-1 bias
    {
        int n = TSTEPS * BATCH * 19;
        k_xt<<<(n + 255) / 256, 256, 0, stream>>>(state, action, conv_w, conv_b, xt);
    }
    {
        int n = TSTEPS * BATCH * FOURH;
        k_pre0<<<(n + 255) / 256, 256, 0, stream>>>(xt, w_ih0, b_ih0, b_hh0, pre0);
    }
    {
        int n = BATCH * FOURH;
        k_pre1<<<(n + 255) / 256, 256, 0, stream>>>(b_ih1, b_hh1, pre1);
    }
    // 4) recurrence: 2 layers x 4 steps; stream order carries the dependency.
    //    Ping-pong h buffers so no workgroup reads h while another writes it.
    for (int t = 0; t < TSTEPS; ++t) {
        __bf16* h0_in  = h0bf + (size_t)(t & 1) * BATCH * H;
        __bf16* h0_out = h0bf + (size_t)((t + 1) & 1) * BATCH * H;
        __bf16* h1_in  = h1bf + (size_t)(t & 1) * BATCH * H;
        __bf16* h1_out = h1bf + (size_t)((t + 1) & 1) * BATCH * H;
        // Layer 0: gates = pre0[t] + h0 @ w_hh0^T
        k_cell<<<H / 16, 128, 0, stream>>>(
            h0_in, wbf_hh0, nullptr, nullptr,
            pre0 + (size_t)t * BATCH * FOURH, c0, nullptr, h0_out);
        // Layer 1: gates = bias1 + h1 @ w_hh1^T + h0_t @ w_ih1^T
        k_cell<<<H / 16, 128, 0, stream>>>(
            h1_in, wbf_hh1, h0_out, wbf_ih1,
            pre1, c1, h1f + (size_t)t * BATCH * H, h1_out);
    }
    // 5) final linear + sigmoid
    k_final<<<BATCH, 256, 0, stream>>>(h1f, lin_w, lin_b, out);
}
